// GraphAttentionLayer_19653770346634
// MI455X (gfx1250) — matrix operand order repaired and checked by reading, not compiled, not run on hardware
//
#include <hip/hip_runtime.h>
#include <hip/hip_bf16.h>

#define N_NODES 8192
#define F_IN    256
#define F_OUT   128
#define ALPHA   0.2f
#define NEGV    (-9.0e15f)

#define CHUNK   128          // j-columns staged per pipeline stage
#define PJ2     136          // padded LDS row stride (bf16 elems): 272B = 17*16B

typedef float  v2f   __attribute__((ext_vector_type(2)));
typedef float  v8f   __attribute__((ext_vector_type(8)));
typedef __bf16 v8bf  __attribute__((ext_vector_type(8)));
typedef __bf16 v16bf __attribute__((ext_vector_type(16)));

// ---------------------------------------------------------------------------
// Kernel 1: Wh = h @ W^T via fp32 WMMA 16x16x4. Also writes bf16 Wh^T for the
// attention pass (B-fragment layout wants contiguous-K at fixed N).
// Block = 256 threads (8 waves). Block -> 16 rows of h; wave w -> 16 cols.
// ---------------------------------------------------------------------------
__global__ __launch_bounds__(256) void wh_gemm_kernel(const float* __restrict__ h,
                                                      const float* __restrict__ W,
                                                      float* __restrict__ Wh,
                                                      __bf16* __restrict__ WhT) {
  const int lane = threadIdx.x & 31;
  const int wave = threadIdx.x >> 5;
  const int m    = lane & 15;
  const int hi   = lane >> 4;     // half-wave select
  const int koff = hi * 2;        // f32 A/B layout: lanes 16-31 hold K+2,K+3
  const int i0   = blockIdx.x * 16;
  const int n0   = wave * 16;

  const float* __restrict__ ha = h + (size_t)(i0 + m) * F_IN;
  const float* __restrict__ wb = W + (size_t)(n0 + m) * F_IN;

  v8f acc = {};
  #pragma unroll 4
  for (int k = 0; k < F_IN; k += 4) {
    v2f av = *(const v2f*)(ha + k + koff);   // A(m, k+koff .. +1)
    v2f bv = *(const v2f*)(wb + k + koff);   // B(k+koff .. +1, n) = W[n][k..]
    acc = __builtin_amdgcn_wmma_f32_16x16x4_f32(false, av, false, bv, (short)0,
                                                acc, false, false);
  }
  const int rbase = hi * 8;  // C layout: VGPR r holds M = r (+8 for hi half)
  #pragma unroll
  for (int r = 0; r < 8; ++r) {
    const int row = i0 + rbase + r;
    const int col = n0 + m;
    const float v = acc[r];
    Wh[(size_t)row * F_OUT + col]    = v;
    WhT[(size_t)col * N_NODES + row] = (__bf16)v;   // contiguous over r -> b128
  }
}

// ---------------------------------------------------------------------------
// Kernel 2: f1 = Wh @ a1, f2 = Wh @ a2. One wave per row (128 = 32 lanes x 4).
// ---------------------------------------------------------------------------
__global__ __launch_bounds__(256) void f1f2_kernel(const float* __restrict__ Wh,
                                                   const float* __restrict__ a,
                                                   float* __restrict__ f1,
                                                   float* __restrict__ f2) {
  const int lane = threadIdx.x & 31;
  const int wave = threadIdx.x >> 5;
  const int row  = blockIdx.x * 8 + wave;
  const float4 w4  = *(const float4*)(Wh + (size_t)row * F_OUT + lane * 4);
  const float4 a14 = *(const float4*)(a + lane * 4);
  const float4 a24 = *(const float4*)(a + F_OUT + lane * 4);
  float s1 = w4.x*a14.x + w4.y*a14.y + w4.z*a14.z + w4.w*a14.w;
  float s2 = w4.x*a24.x + w4.y*a24.y + w4.z*a24.z + w4.w*a24.w;
  #pragma unroll
  for (int d = 16; d >= 1; d >>= 1) {
    s1 += __shfl_xor(s1, d, 32);
    s2 += __shfl_xor(s2, d, 32);
  }
  if (lane == 0) { f1[row] = s1; f2[row] = s2; }
}

// ---------------------------------------------------------------------------
// Kernel 3: per-row softmax stats. Reads the 268MB adj stream ONCE (coalesced
// int4), emits rowmax, 1/rowsum, and a 1-bit/edge mask (8MB) so the attention
// pass never re-reads adj. Block per row; thread t owns 32 consecutive j.
// ---------------------------------------------------------------------------
__global__ __launch_bounds__(256) void rowstats_kernel(const int* __restrict__ adj,
                                                       const float* __restrict__ f1,
                                                       const float* __restrict__ f2,
                                                       float* __restrict__ rmax,
                                                       float* __restrict__ rinv,
                                                       unsigned int* __restrict__ mask) {
  __shared__ float red[8];
  const int tid  = threadIdx.x;
  const int lane = tid & 31;
  const int wave = tid >> 5;
  const int row  = blockIdx.x;
  const float f1r = f1[row];
  const int jbase = tid * 32;
  const int* __restrict__ arow = adj + (size_t)row * N_NODES + jbase;

  float s[32];
  unsigned int mword = 0;
  float tmax = NEGV;
  #pragma unroll
  for (int q = 0; q < 8; ++q) {
    const int4   av = *(const int4*)(arow + q * 4);
    const float4 fv = *(const float4*)(f2 + jbase + q * 4);
    const int   am[4] = {av.x, av.y, av.z, av.w};
    const float fm[4] = {fv.x, fv.y, fv.z, fv.w};
    #pragma unroll
    for (int b = 0; b < 4; ++b) {
      const float x = f1r + fm[b];
      const float e = fmaxf(x, ALPHA * x);   // leaky_relu (ALPHA < 1)
      const bool on = am[b] > 0;
      const float sv = on ? e : NEGV;
      s[q * 4 + b] = sv;
      tmax = fmaxf(tmax, sv);
      if (on) mword |= (1u << (q * 4 + b));
    }
  }
  mask[(size_t)row * (N_NODES / 32) + tid] = mword;

  #pragma unroll
  for (int d = 16; d >= 1; d >>= 1) tmax = fmaxf(tmax, __shfl_xor(tmax, d, 32));
  if (lane == 0) red[wave] = tmax;
  __syncthreads();
  float bmax = red[0];
  #pragma unroll
  for (int w = 1; w < 8; ++w) bmax = fmaxf(bmax, red[w]);
  __syncthreads();

  float lsum = 0.f;
  #pragma unroll
  for (int q = 0; q < 32; ++q) lsum += __expf(s[q] - bmax);
  #pragma unroll
  for (int d = 16; d >= 1; d >>= 1) lsum += __shfl_xor(lsum, d, 32);
  if (lane == 0) red[wave] = lsum;
  __syncthreads();
  if (tid == 0) {
    float tot = 0.f;
    #pragma unroll
    for (int w = 0; w < 8; ++w) tot += red[w];
    rmax[row] = bmax;
    rinv[row] = 1.0f / tot;
  }
}

// ---------------------------------------------------------------------------
// Kernel 4: out = softmax(scores) @ Wh via bf16 WMMA 16x16x32 (f32 accum).
// Block -> 16 output rows; wave w -> output cols [16w, 16w+16).
// Pipelined: the normalized P tile (pre-scaled by 1/rowsum, computed once per
// block, bf16 in LDS) is DOUBLE-BUFFERED so staging of chunk i+1 overlaps the
// 4 WMMAs of chunk i -> one barrier per 128-j chunk (64 barriers total).
// The accumulator is the final output (P already normalized).
// ---------------------------------------------------------------------------
__global__ __launch_bounds__(256) void attn_kernel(const unsigned int* __restrict__ mask,
                                                   const __bf16* __restrict__ WhT,
                                                   const float* __restrict__ f1,
                                                   const float* __restrict__ f2,
                                                   const float* __restrict__ rmax,
                                                   const float* __restrict__ rinv,
                                                   float* __restrict__ out) {
  __shared__ __bf16 s_p[2][16 * PJ2];   // 2 x 16 x 136 bf16 = 8704 B

  const int tid  = threadIdx.x;
  const int lane = tid & 31;
  const int wave = tid >> 5;
  const int m    = lane & 15;
  const int hi   = lane >> 4;
  const int i0   = blockIdx.x * 16;
  const int n0   = wave * 16;

  // P-staging role: thread t -> row (t&15), 8 consecutive j at 8*(t>>4)
  const int pm  = tid & 15;
  const int jq8 = (tid >> 4) * 8;
  const float f1r = f1[i0 + pm];
  const float mr  = rmax[i0 + pm];
  const float iv  = rinv[i0 + pm];
  const unsigned int* __restrict__ mrow = mask + (size_t)(i0 + pm) * (N_NODES / 32);
  const int bit0 = jq8 & 31;           // {0,8,16,24}

  // B base: Wh^T row for this lane's output column; K(e) = 16*hi + e
  const __bf16* __restrict__ bp = WhT + (size_t)(n0 + m) * N_NODES + hi * 16;

  auto stage = [&](int jt, __bf16* buf) {
    const unsigned int mw = mrow[(jt + jq8) >> 5];
    const float4 f2a = *(const float4*)(f2 + jt + jq8);
    const float4 f2b = *(const float4*)(f2 + jt + jq8 + 4);
    const float fm[8] = {f2a.x, f2a.y, f2a.z, f2a.w, f2b.x, f2b.y, f2b.z, f2b.w};
    v8bf pv;
    #pragma unroll
    for (int b = 0; b < 8; ++b) {
      const float x  = f1r + fm[b];
      const float e  = fmaxf(x, ALPHA * x);
      const float sv = ((mw >> (bit0 + b)) & 1u) ? e : NEGV;
      pv[b] = (__bf16)(__expf(sv - mr) * iv);
    }
    *(v8bf*)&buf[pm * PJ2 + jq8] = pv;   // 16B aligned: 272*pm + 2*jq8
  };

  v8f acc = {};
  stage(0, s_p[0]);

  for (int jt = 0; jt < N_NODES; jt += CHUNK) {
    const int cur = (jt >> 7) & 1;
    __syncthreads();                       // staging of `cur` complete;
                                           // readers of cur^1 are done
    if (jt + CHUNK < N_NODES) {
      stage(jt + CHUNK, s_p[cur ^ 1]);     // overlap next stage with WMMAs
      __builtin_prefetch(bp + jt + CHUNK, 0, 1);  // -> global_prefetch_b8
    }

    const __bf16* pbase = &s_p[cur][m * PJ2 + hi * 16];
    #pragma unroll
    for (int g = 0; g < 4; ++g) {          // four K=32 WMMA steps per chunk
      const int jb = g * 32;
      // A fragment: contiguous K = 16*hi + (0..15), matching B
      const v8bf alo = *(const v8bf*)(pbase + jb);        // K = 16hi + 0..7
      const v8bf ahi = *(const v8bf*)(pbase + jb + 8);    // K = 16hi+8+0..7
      const v16bf av = __builtin_shufflevector(alo, ahi,
          0, 1, 2, 3, 4, 5, 6, 7, 8, 9, 10, 11, 12, 13, 14, 15);
      // B fragment: contiguous 16 bf16 (32B aligned) from Wh^T
      const v16bf bv = *(const v16bf*)(bp + jt + jb);
      acc = __builtin_amdgcn_wmma_f32_16x16x32_bf16(false, av, false, bv,
                                                    (short)0, acc, false, false);
    }
  }

  #pragma unroll
  for (int r = 0; r < 8; ++r)
    out[(size_t)(i0 + hi * 8 + r) * F_OUT + n0 + m] = acc[r];
}

// ---------------------------------------------------------------------------
extern "C" void kernel_launch(void* const* d_in, const int* in_sizes, int n_in,
                              void* d_out, int out_size, void* d_ws, size_t ws_size,
                              hipStream_t stream) {
  (void)in_sizes; (void)n_in; (void)out_size; (void)ws_size;
  const float* h   = (const float*)d_in[0];
  const int*   adj = (const int*)d_in[1];
  const float* W   = (const float*)d_in[2];
  const float* a   = (const float*)d_in[3];
  float* out = (float*)d_out;

  char* ws = (char*)d_ws;
  float* Wh = (float*)ws;                 ws += (size_t)N_NODES * F_OUT * sizeof(float);
  __bf16* WhT = (__bf16*)ws;              ws += (size_t)F_OUT * N_NODES * sizeof(__bf16);
  float* f1 = (float*)ws;                 ws += (size_t)N_NODES * sizeof(float);
  float* f2 = (float*)ws;                 ws += (size_t)N_NODES * sizeof(float);
  float* rmx = (float*)ws;                ws += (size_t)N_NODES * sizeof(float);
  float* rin = (float*)ws;                ws += (size_t)N_NODES * sizeof(float);
  unsigned int* mask = (unsigned int*)ws; ws += (size_t)N_NODES * (N_NODES / 32) * 4;

  wh_gemm_kernel<<<N_NODES / 16, 256, 0, stream>>>(h, W, Wh, WhT);
  f1f2_kernel<<<N_NODES / 8, 256, 0, stream>>>(Wh, a, f1, f2);
  rowstats_kernel<<<N_NODES, 256, 0, stream>>>(adj, f1, f2, rmx, rin, mask);
  attn_kernel<<<N_NODES / 16, 256, 0, stream>>>(mask, WhT, f1, f2, rmx, rin, out);
}
